// Model_61512521613593
// MI455X (gfx1250) — matrix operand order repaired and checked by reading, not compiled, not run on hardware
//
#include <hip/hip_runtime.h>

typedef _Float16 f16;
typedef __attribute__((ext_vector_type(16))) _Float16 v16h;
typedef __attribute__((ext_vector_type(8)))  float    v8f;

#define WMMA_F16(a, b, c) \
  __builtin_amdgcn_wmma_f32_16x16x32_f16(false, (a), false, (b), (short)0, (c), false, false)

__device__ __forceinline__ float sigm(float x)  { return 1.0f / (1.0f + __expf(-x)); }
__device__ __forceinline__ float tanh_(float x) { return 1.0f - 2.0f / (__expf(2.0f * x) + 1.0f); }

// ---------------------------------------------------------------------------
// Kernel 1: encoder linear (37 -> 64) + ReLU, output f16 laid out [T][B][64]
// so the recurrent kernel reads a contiguous 2KB tile per (t, batch-tile).
// ---------------------------------------------------------------------------
__global__ __launch_bounds__(256) void k_enc_linear(
    const float* __restrict__ x, const float* __restrict__ W,
    const float* __restrict__ bias, f16* __restrict__ e16, int B, int T)
{
  __shared__ float Wl[64 * 37];
  __shared__ float bl[64];
  __shared__ float xl[4][37];
  const int tid = threadIdx.x;
  for (int i = tid; i < 64 * 37; i += 256) Wl[i] = W[i];
  if (tid < 64) bl[tid] = bias[tid];
  const int bt0 = blockIdx.x * 4;
  for (int i = tid; i < 4 * 37; i += 256) {
    int l = i / 37, j = i % 37;
    xl[l][j] = x[(size_t)(bt0 + l) * 37 + j];
  }
  __syncthreads();
  const int l = tid >> 6;     // which of the 4 (b,t) pairs
  const int k = tid & 63;     // output feature
  const int bt = bt0 + l;
  const int b = bt / T, t = bt % T;
  float acc = bl[k];
  #pragma unroll
  for (int j = 0; j < 37; ++j) acc += xl[l][j] * Wl[k * 37 + j];
  acc = acc > 0.0f ? acc : 0.0f;
  e16[((size_t)t * B + b) * 64 + k] = (f16)acc;
}

// ---------------------------------------------------------------------------
// Kernel 2: fused bidirectional LSTM + running sum over t (the "attention"
// context fd — softmax over a singleton axis is exactly 1.0, so
// fd[b,:] = sum_t feats[b,t,:], and feats never needs materializing).
// grid = (B/16, 2 directions), block = 128 threads (4 waves; wave g = gate g).
// Weights held in VGPRs as WMMA B-operands; h/c/gates staged through LDS.
// ---------------------------------------------------------------------------
__global__ __launch_bounds__(128) void k_bilstm(
    const f16* __restrict__ e16,
    const float* __restrict__ wih_f, const float* __restrict__ whh_f,
    const float* __restrict__ bih_f, const float* __restrict__ bhh_f,
    const float* __restrict__ wih_b, const float* __restrict__ whh_b,
    const float* __restrict__ bih_b, const float* __restrict__ bhh_b,
    float* __restrict__ fd, int B, int T)
{
  __shared__ float gl[4][16][64];   // gates i,f,g,o (f32 WMMA accumulators)
  __shared__ f16   h16[16][64];     // h as WMMA A-source
  __shared__ float cst[16][64];     // cell state (f32)
  __shared__ float fda[16][64];     // running sum of h over t
  __shared__ float bsum[256];       // bih + bhh

  const int dir  = blockIdx.y;
  const int b0   = blockIdx.x * 16;
  const int tid  = threadIdx.x;
  const int wave = tid >> 5, lane = tid & 31;
  const int nl   = lane & 15;       // N col (B/C/D) == M row (A)
  const int kh   = lane >> 4;       // lane half

  const float* wih = dir ? wih_b : wih_f;
  const float* whh = dir ? whh_b : whh_f;
  const float* bih = dir ? bih_b : bih_f;
  const float* bhh = dir ? bhh_b : bhh_f;

  for (int i = tid; i < 16 * 64; i += 128) {
    int m = i >> 6, j = i & 63;
    h16[m][j] = (f16)0.0f; cst[m][j] = 0.0f; fda[m][j] = 0.0f;
  }
  for (int i = tid; i < 256; i += 128) bsum[i] = bih[i] + bhh[i];

  // Resident B-operands. B[k][n] = w[n*64 + k]; lane layout: n = nl, k = i + 16*kh.
  v16h Bx[2][4], Bh[2][4];
  const int n0 = wave * 64;
  #pragma unroll
  for (int kt = 0; kt < 2; ++kt)
    #pragma unroll
    for (int nt = 0; nt < 4; ++nt) {
      const int n = n0 + nt * 16 + nl;
      const int kb = kt * 32 + kh * 16;
      #pragma unroll
      for (int i = 0; i < 16; ++i) {
        Bx[kt][nt][i] = (f16)wih[n * 64 + kb + i];
        Bh[kt][nt][i] = (f16)whh[n * 64 + kb + i];
      }
    }
  __syncthreads();

  const int koffA = kh * 16;  // A layout: k = i + 16*(lane/16)
  for (int s = 0; s < T; ++s) {
    const int t = dir ? (T - 1 - s) : s;
    const f16* erow = e16 + ((size_t)t * B + b0 + nl) * 64;
    v16h ax0, ax1, ah0, ah1;
    #pragma unroll
    for (int i = 0; i < 16; ++i) {
      const int k = i + koffA;
      ax0[i] = erow[k];
      ax1[i] = erow[k + 32];
      ah0[i] = h16[nl][k];
      ah1[i] = h16[nl][k + 32];
    }
    #pragma unroll
    for (int nt = 0; nt < 4; ++nt) {
      v8f acc = {0.f, 0.f, 0.f, 0.f, 0.f, 0.f, 0.f, 0.f};
      acc = WMMA_F16(ax0, Bx[0][nt], acc);
      acc = WMMA_F16(ax1, Bx[1][nt], acc);
      acc = WMMA_F16(ah0, Bh[0][nt], acc);
      acc = WMMA_F16(ah1, Bh[1][nt], acc);
      #pragma unroll
      for (int v = 0; v < 8; ++v)
        gl[wave][v + kh * 8][nt * 16 + nl] = acc[v];
    }
    __syncthreads();
    #pragma unroll
    for (int r = 0; r < 8; ++r) {
      const int idx = r * 128 + tid;
      const int m = idx >> 6, j = idx & 63;
      const float gi = gl[0][m][j] + bsum[j];
      const float gf = gl[1][m][j] + bsum[64 + j];
      const float gg = gl[2][m][j] + bsum[128 + j];
      const float go = gl[3][m][j] + bsum[192 + j];
      const float c  = sigm(gf) * cst[m][j] + sigm(gi) * tanh_(gg);
      const float h  = sigm(go) * tanh_(c);
      cst[m][j] = c;
      h16[m][j] = (f16)h;
      fda[m][j] += h;
    }
    __syncthreads();
  }
  for (int i = tid; i < 16 * 64; i += 128) {
    const int m = i >> 6, j = i & 63;
    fd[(size_t)(b0 + m) * 128 + dir * 64 + j] = fda[m][j];
  }
}

// ---------------------------------------------------------------------------
// Kernel 3: decoder. fd is constant across steps, so Gq = fd@dec_wih^T + bias
// is computed once (WMMA, K=128), then 10 recurrent steps of h@dec_whh^T
// (WMMA, K=64) + LSTM cell + output linear/softmax.
// ---------------------------------------------------------------------------
__global__ __launch_bounds__(128) void k_decoder(
    const float* __restrict__ fd,
    const float* __restrict__ dwih, const float* __restrict__ dwhh,
    const float* __restrict__ dbih, const float* __restrict__ dbhh,
    const float* __restrict__ dlw,  const float* __restrict__ dlb,
    float* __restrict__ out, int B, int L)
{
  __shared__ float Gq[4][16][64];
  __shared__ float gl[4][16][64];
  __shared__ f16   h16[16][64];
  __shared__ float hF[16][64];
  __shared__ float cst[16][64];
  __shared__ float bsum[256];

  const int b0   = blockIdx.x * 16;
  const int tid  = threadIdx.x;
  const int wave = tid >> 5, lane = tid & 31;
  const int nl   = lane & 15, kh = lane >> 4;

  for (int i = tid; i < 16 * 64; i += 128) {
    int m = i >> 6, j = i & 63;
    h16[m][j] = (f16)0.0f; hF[m][j] = 0.0f; cst[m][j] = 0.0f;
  }
  for (int i = tid; i < 256; i += 128) bsum[i] = dbih[i] + dbhh[i];
  __syncthreads();

  const int n0 = wave * 64;
  v16h Bw[4][4], Bh[2][4];
  #pragma unroll
  for (int kt = 0; kt < 4; ++kt)
    #pragma unroll
    for (int nt = 0; nt < 4; ++nt) {
      const int n = n0 + nt * 16 + nl;
      const int kb = kt * 32 + kh * 16;
      #pragma unroll
      for (int i = 0; i < 16; ++i)
        Bw[kt][nt][i] = (f16)dwih[n * 128 + kb + i];
    }
  #pragma unroll
  for (int kt = 0; kt < 2; ++kt)
    #pragma unroll
    for (int nt = 0; nt < 4; ++nt) {
      const int n = n0 + nt * 16 + nl;
      const int kb = kt * 32 + kh * 16;
      #pragma unroll
      for (int i = 0; i < 16; ++i)
        Bh[kt][nt][i] = (f16)dwhh[n * 64 + kb + i];
    }

  const int koffA = kh * 16;
  const float* frow = fd + (size_t)(b0 + nl) * 128;
  v16h af[4];
  #pragma unroll
  for (int kt = 0; kt < 4; ++kt)
    #pragma unroll
    for (int i = 0; i < 16; ++i) {
      const int k = i + koffA + kt * 32;
      af[kt][i] = (f16)frow[k];
    }

  #pragma unroll
  for (int nt = 0; nt < 4; ++nt) {
    v8f acc = {0.f, 0.f, 0.f, 0.f, 0.f, 0.f, 0.f, 0.f};
    #pragma unroll
    for (int kt = 0; kt < 4; ++kt) acc = WMMA_F16(af[kt], Bw[kt][nt], acc);
    #pragma unroll
    for (int v = 0; v < 8; ++v)
      Gq[wave][v + kh * 8][nt * 16 + nl] = acc[v] + bsum[n0 + nt * 16 + nl];
  }
  __syncthreads();

  for (int s = 0; s < L; ++s) {
    v16h ah0, ah1;
    #pragma unroll
    for (int i = 0; i < 16; ++i) {
      const int k = i + koffA;
      ah0[i] = h16[nl][k];
      ah1[i] = h16[nl][k + 32];
    }
    #pragma unroll
    for (int nt = 0; nt < 4; ++nt) {
      v8f acc;
      #pragma unroll
      for (int v = 0; v < 8; ++v) acc[v] = Gq[wave][v + kh * 8][nt * 16 + nl];
      acc = WMMA_F16(ah0, Bh[0][nt], acc);
      acc = WMMA_F16(ah1, Bh[1][nt], acc);
      #pragma unroll
      for (int v = 0; v < 8; ++v)
        gl[wave][v + kh * 8][nt * 16 + nl] = acc[v];
    }
    __syncthreads();
    #pragma unroll
    for (int r = 0; r < 8; ++r) {
      const int idx = r * 128 + tid;
      const int m = idx >> 6, j = idx & 63;
      const float c = sigm(gl[1][m][j]) * cst[m][j] + sigm(gl[0][m][j]) * tanh_(gl[2][m][j]);
      const float h = sigm(gl[3][m][j]) * tanh_(c);
      cst[m][j] = c;
      h16[m][j] = (f16)h;
      hF[m][j]  = h;
    }
    __syncthreads();
    if (tid < 16) {
      const int m = tid;
      float logit[11];
      float mx = -1e30f;
      #pragma unroll
      for (int q = 0; q < 11; ++q) {
        float a = dlb[q];
        #pragma unroll
        for (int k = 0; k < 64; ++k) a += hF[m][k] * dlw[q * 64 + k];
        logit[q] = a;
        mx = fmaxf(mx, a);
      }
      float sum = 0.0f;
      #pragma unroll
      for (int q = 0; q < 11; ++q) { logit[q] = __expf(logit[q] - mx); sum += logit[q]; }
      const float inv = 1.0f / sum;
      const size_t base = ((size_t)(b0 + m) * L + s) * 11;
      #pragma unroll
      for (int q = 0; q < 11; ++q) out[base + q] = logit[q] * inv;
    }
    __syncthreads();
  }
}

// ---------------------------------------------------------------------------
extern "C" void kernel_launch(void* const* d_in, const int* in_sizes, int n_in,
                              void* d_out, int out_size, void* d_ws, size_t ws_size,
                              hipStream_t stream) {
  const float* x         = (const float*)d_in[0];
  const float* enc_lin_w = (const float*)d_in[1];
  const float* enc_lin_b = (const float*)d_in[2];
  const float* wih_f     = (const float*)d_in[3];
  const float* whh_f     = (const float*)d_in[4];
  const float* bih_f     = (const float*)d_in[5];
  const float* bhh_f     = (const float*)d_in[6];
  const float* wih_b     = (const float*)d_in[7];
  const float* whh_b     = (const float*)d_in[8];
  const float* bih_b     = (const float*)d_in[9];
  const float* bhh_b     = (const float*)d_in[10];
  // d_in[11..14]: attention weights — provably unused (softmax over a
  // singleton axis is identically 1.0, so fd = sum_t feats).
  const float* dec_wih   = (const float*)d_in[15];
  const float* dec_whh   = (const float*)d_in[16];
  const float* dec_bih   = (const float*)d_in[17];
  const float* dec_bhh   = (const float*)d_in[18];
  const float* dec_lin_w = (const float*)d_in[19];
  const float* dec_lin_b = (const float*)d_in[20];

  const int T = 128;
  const int B = in_sizes[0] / (37 * T);       // 2048
  const int L = out_size / (B * 11);          // 10

  char* ws = (char*)d_ws;
  f16*   e16 = (f16*)ws;
  size_t e_bytes = (size_t)B * T * 64 * sizeof(f16);
  float* fd = (float*)(ws + ((e_bytes + 255) & ~(size_t)255));
  (void)n_in; (void)ws_size;

  k_enc_linear<<<(B * T) / 4, 256, 0, stream>>>(x, enc_lin_w, enc_lin_b, e16, B, T);
  dim3 g2(B / 16, 2);
  k_bilstm<<<g2, 128, 0, stream>>>(e16, wih_f, whh_f, bih_f, bhh_f,
                                   wih_b, whh_b, bih_b, bhh_b, fd, B, T);
  k_decoder<<<B / 16, 128, 0, stream>>>(fd, dec_wih, dec_whh, dec_bih, dec_bhh,
                                        dec_lin_w, dec_lin_b, (float*)d_out, B, L);
}
